// Kernel_17164586276825551330_71605694759419
// MI455X (gfx1250) — matrix operand order repaired and checked by reading, not compiled, not run on hardware
//
#include <hip/hip_runtime.h>

// Problem dims
#define NB 32
#define CC 128
#define HH 56
#define WW 56
#define GG 8

typedef __attribute__((ext_vector_type(16))) __bf16 v16bf;
typedef __attribute__((ext_vector_type(8)))  float  v8f;
typedef __attribute__((vector_size(16)))     int    v4i_t;   // payload type for async b128

// Workspace layout (bytes)
#define A1_BYTES (8*5*4*32*32)        // 163840: w1 fragments, [coblk8][kh5][cb4][lane32][16 bf16]
#define A4_BYTES (3*4*32*32)          // 12288 : w4 fragments, [tap3][cb4][lane32][16 bf16]
#define A1_ELEMS (A1_BYTES/2)
#define XB_OFF_ELEMS ((A1_BYTES + A4_BYTES)/2)  // x in bf16, [n][h][w][ci]

#if defined(__HIP_DEVICE_COMPILE__) && __has_builtin(__builtin_amdgcn_global_load_async_to_lds_b128)
#define USE_ASYNC_LDS 1
#endif

__device__ __forceinline__ unsigned short f2bf(float f) {
  unsigned int u = __float_as_uint(f);
  unsigned int r = u + 0x7fffu + ((u >> 16) & 1u);   // round-to-nearest-even
  return (unsigned short)(r >> 16);
}

union FragU {
  uint4 q[2];
  v16bf v;
};

// ---------------------------------------------------------------------------
// Kernel 1: build pre-swizzled bf16 A-fragments for w1 and w4.
// A 16x32 bf16 fragment layout (ISA 7.12.2): lanes 0-15: K0..7 in v0..3,
// K16..23 in v4..7; lanes 16-31: +8 on K.
// ---------------------------------------------------------------------------
__global__ void prep_weights(const float* __restrict__ w1,
                             const float* __restrict__ w4,
                             unsigned short* __restrict__ wsb) {
  const int TOT1 = 8 * 5 * 4 * 512;  // 81920 elements
  const int TOT2 = 3 * 4 * 512;      // 6144 elements
  for (int i = blockIdx.x * blockDim.x + threadIdx.x; i < TOT1 + TOT2;
       i += gridDim.x * blockDim.x) {
    if (i < TOT1) {
      int e = i & 15, lane = (i >> 4) & 31, fid = i >> 9;
      int cb = fid & 3, kh = (fid >> 2) % 5, coblk = fid / 20;
      int m = lane & 15;
      int v = e >> 1, hb = e & 1;
      int k = (v < 4 ? 2 * v + hb : 2 * (v - 4) + hb + 16) + ((lane >= 16) ? 8 : 0);
      int co = coblk * 16 + m, ci = cb * 32 + k;
      wsb[i] = f2bf(w1[(co * CC + ci) * 5 + kh]);      // w1: (C,C,5,1)
    } else {
      int i2 = i - TOT1;
      int e = i2 & 15, lane = (i2 >> 4) & 31, fid = i2 >> 9;
      int cb = fid & 3, tap = fid >> 2;
      int m = lane & 15;
      int v = e >> 1, hb = e & 1;
      int k = (v < 4 ? 2 * v + hb : 2 * (v - 4) + hb + 16) + ((lane >= 16) ? 8 : 0);
      int ci = cb * 32 + k;
      float val = (m < 8) ? w4[ci * 24 + tap * 8 + m] : 0.0f;  // w4: (C,3,1,G); rows 8..15 zero-pad
      wsb[i] = f2bf(val);
    }
  }
}

// ---------------------------------------------------------------------------
// Kernel 2: x fp32 NCHW -> bf16 [n][h][w][ci]  (channel-contiguous for B frags)
// One block per (n,h); transpose through LDS so both sides are coalesced.
// ---------------------------------------------------------------------------
__global__ __launch_bounds__(256) void xpose(const float* __restrict__ x,
                                             unsigned short* __restrict__ xb) {
  __shared__ unsigned short tile[CC * WW];
  const int nh = blockIdx.x;
  const int n = nh / HH, h = nh % HH;
  for (int i = threadIdx.x; i < CC * WW; i += 256) {
    int ci = i / WW, w = i % WW;                      // read coalesced along w
    tile[i] = f2bf(x[((size_t)(n * CC + ci) * HH + h) * WW + w]);
  }
  __syncthreads();
  for (int i = threadIdx.x; i < WW * CC; i += 256) {
    int ci = i & 127, w = i >> 7;                     // write coalesced along ci
    xb[(size_t)(nh * WW + w) * CC + ci] = tile[ci * WW + w];
  }
}

// ---------------------------------------------------------------------------
// Kernel 3: fused conv(5,1) -> depthwise(3,1) * gate einsum, all WMMA bf16.
// Block = (n, 4 h rows, 16 w cols). 8 waves, wave = 16 output channels.
// ---------------------------------------------------------------------------
__global__ __launch_bounds__(256) void fused_conv(const unsigned short* __restrict__ wsb,
                                                  const float* __restrict__ w3,
                                                  float* __restrict__ out) {
  __shared__ __align__(16) unsigned short xtile[10 * 20 * 128];  // 51200 B, [hp*20+wp][ci] bf16
  __shared__ float t4buf[4 * 8 * 16];                            // [j][g][w]

  const int tid = threadIdx.x;
  const int w0 = (blockIdx.x < 3) ? (int)blockIdx.x * 16 : 40;   // covers 0..55, tile 3 overlaps
  const int h0 = (int)blockIdx.y * 4;
  const int n = (int)blockIdx.z;
  const unsigned short* xb = wsb + XB_OFF_ELEMS;

  // Zero-pad halo only for boundary tiles.
  bool boundary = (h0 == 0) || (h0 == HH - 4) || (w0 == 0) || (w0 == 40);
  if (boundary) {
    uint4 z = make_uint4(0u, 0u, 0u, 0u);
    for (int i = tid; i < 3200; i += 256) reinterpret_cast<uint4*>(xtile)[i] = z;
    __syncthreads();
  }

  // Stage x tile: rows (hp,wp) are 256 B contiguous in global -> 16 b128 xfers each.
  for (int t = tid; t < 3200; t += 256) {
    int row = t >> 4, seg = t & 15;
    int hp = row / 20, wp = row - hp * 20;
    int xh = h0 - 3 + hp, xw = w0 - 2 + wp;
    if (((unsigned)xh < (unsigned)HH) && ((unsigned)xw < (unsigned)WW)) {
      const unsigned short* gp = xb + (size_t)((n * HH + xh) * WW + xw) * CC + seg * 8;
      unsigned short* lp = xtile + row * 128 + seg * 8;
#ifdef USE_ASYNC_LDS
      __builtin_amdgcn_global_load_async_to_lds_b128(
          (__attribute__((address_space(1))) v4i_t*)gp,
          (__attribute__((address_space(3))) v4i_t*)lp, 0, 0);
#else
      *reinterpret_cast<uint4*>(lp) = *reinterpret_cast<const uint4*>(gp);
#endif
    }
  }
#ifdef USE_ASYNC_LDS
#if __has_builtin(__builtin_amdgcn_s_wait_asynccnt)
  __builtin_amdgcn_s_wait_asynccnt(0);
#else
  asm volatile("s_wait_asynccnt 0" ::: "memory");
#endif
#endif
  __syncthreads();

  const int wave = tid >> 5;          // wave32
  const int lane = tid & 31;
  const int col = lane & 15;          // B/C/D column (w offset in tile)
  const int cihalf = (lane & 16) >> 1; // +8 ci for high half-wave (B layout)
  const int mo = (lane & 16) >> 1;    // C/D row offset: lanes>=16 hold M = r+8
  const int co_base = wave * 16;

  // ---- t1: 16co x 16w x 6 h-rows, K = 5 taps x 128 ci (20 WMMAs per row) ----
  v8f acc[6] = {};
#pragma unroll
  for (int cb = 0; cb < 4; ++cb) {
    v16bf A[5];
#pragma unroll
    for (int kh = 0; kh < 5; ++kh) {
      const uint4* pa = reinterpret_cast<const uint4*>(
          wsb + (size_t)(((wave * 5 + kh) * 4 + cb)) * 512 + lane * 16);
      FragU u; u.q[0] = pa[0]; u.q[1] = pa[1];
      A[kh] = u.v;
    }
#pragma unroll
    for (int row = 0; row < 10; ++row) {
      int bo = (row * 20 + col + 2) * 128 + cb * 32 + cihalf;
      FragU u;
      u.q[0] = *reinterpret_cast<const uint4*>(&xtile[bo]);
      u.q[1] = *reinterpret_cast<const uint4*>(&xtile[bo + 16]);
      v16bf B = u.v;
#pragma unroll
      for (int kh = 0; kh < 5; ++kh) {
        int hh = row - kh;                       // accumulator h' = h0-1+hh
        if (hh >= 0 && hh < 6)
          acc[hh] = __builtin_amdgcn_wmma_f32_16x16x32_bf16(
              false, A[kh], false, B, (short)0, acc[hh], false, false);
      }
    }
  }

  // ---- t4 gates: waves 0..3 each do one h row (g x w = 16x16, K = 3 taps x 128 ci) ----
  if (wave < 4) {
    const int j = wave;
    v8f a4 = {};
#pragma unroll
    for (int t = 0; t < 3; ++t) {
#pragma unroll
      for (int cb = 0; cb < 4; ++cb) {
        const uint4* pa = reinterpret_cast<const uint4*>(
            wsb + A1_ELEMS + (size_t)(t * 4 + cb) * 512 + lane * 16);
        FragU ua; ua.q[0] = pa[0]; ua.q[1] = pa[1];
        int bo = (((j + 3) * 20) + col + 2 * t) * 128 + cb * 32 + cihalf;  // w taps -2,0,+2
        FragU ub;
        ub.q[0] = *reinterpret_cast<const uint4*>(&xtile[bo]);
        ub.q[1] = *reinterpret_cast<const uint4*>(&xtile[bo + 16]);
        a4 = __builtin_amdgcn_wmma_f32_16x16x32_bf16(
            false, ua.v, false, ub.v, (short)0, a4, false, false);
      }
    }
    if (lane < 16) {                  // rows m=0..7 live in lanes 0..15, VGPR r -> g=r
#pragma unroll
      for (int r = 0; r < 8; ++r) t4buf[(j * 8 + r) * 16 + lane] = a4[r];
    }
  }
  __syncthreads();

  // ---- t3 (depthwise 3-tap in h, per-channel weights) fused with gate product ----
  float w3a[8], w3b[8], w3c[8];
#pragma unroll
  for (int r = 0; r < 8; ++r) {
    int c = co_base + mo + r;         // channel for VGPR row r in this lane-half
    w3a[r] = w3[c * 3 + 0];
    w3b[r] = w3[c * 3 + 1];
    w3c[r] = w3[c * 3 + 2];
  }
#pragma unroll
  for (int j = 0; j < 4; ++j) {
    size_t obase = ((size_t)(n * CC + co_base + mo) * HH + (h0 + j)) * WW + w0 + col;
#pragma unroll
    for (int r = 0; r < 8; ++r) {
      float t3v = w3a[r] * acc[j][r] + w3b[r] * acc[j + 1][r] + w3c[r] * acc[j + 2][r];
      float t4v = t4buf[(j * 8 + r) * 16 + col];   // g = (c mod 8) = r
      out[obase + (size_t)r * (HH * WW)] = t3v * t4v;
    }
  }
}

// ---------------------------------------------------------------------------
extern "C" void kernel_launch(void* const* d_in, const int* in_sizes, int n_in,
                              void* d_out, int out_size, void* d_ws, size_t ws_size,
                              hipStream_t stream) {
  const float* x  = (const float*)d_in[0];
  const float* w1 = (const float*)d_in[1];
  const float* w3 = (const float*)d_in[2];
  const float* w4 = (const float*)d_in[3];
  float* out = (float*)d_out;
  unsigned short* wsb = (unsigned short*)d_ws;

  prep_weights<<<344, 256, 0, stream>>>(w1, w4, wsb);
  xpose<<<NB * HH, 256, 0, stream>>>(x, wsb + XB_OFF_ELEMS);
  fused_conv<<<dim3(4, 14, NB), 256, 0, stream>>>(wsb, w3, out);
}
